// GraphSAGEFraudDetector_34660386078847
// MI455X (gfx1250) — matrix operand order repaired and checked by reading, not compiled, not run on hardware
//
#include <hip/hip_runtime.h>
#include <hip/hip_bf16.h>

#define BN_EPS 1e-5f

typedef __attribute__((ext_vector_type(16))) __bf16 bf16x16;
typedef __attribute__((ext_vector_type(8)))  float  f32x8;
typedef int v4i_vs __attribute__((vector_size(16)));

union BVec { uint4 q[2]; bf16x16 v; };

#if defined(__has_builtin)
#if __has_builtin(__builtin_amdgcn_global_load_async_to_lds_b128) && \
    __has_builtin(__builtin_amdgcn_s_wait_asynccnt)
#define HAVE_ASYNC_LDS 1
#endif
#endif
#ifndef HAVE_ASYNC_LDS
#define HAVE_ASYNC_LDS 0
#endif

#if HAVE_ASYNC_LDS
typedef __attribute__((address_space(1))) v4i_vs* async_gptr;
typedef __attribute__((address_space(3))) v4i_vs* async_lptr;
#endif

// ---------------- utility kernels ----------------

__global__ void zero_f32_kernel(float* __restrict__ p, long n) {
    long i = (long)blockIdx.x * blockDim.x + threadIdx.x;
    if (i < n) p[i] = 0.0f;
}

__global__ void f32_to_bf16_kernel(const float* __restrict__ in,
                                   __hip_bfloat16* __restrict__ out, long n) {
    long i = (long)blockIdx.x * blockDim.x + threadIdx.x;
    if (i < n) out[i] = __float2bfloat16(in[i]);
}

__global__ void degree_kernel(const int* __restrict__ dst,
                              float* __restrict__ deg, int E) {
    int e = blockIdx.x * blockDim.x + threadIdx.x;
    if (e < E) atomicAdd(&deg[dst[e]], 1.0f);
}

// one edge per wave32; lanes cover the feature dimension (din/32 elems each)
__global__ void scatter_kernel(const __hip_bfloat16* __restrict__ x,
                               const int* __restrict__ src,
                               const int* __restrict__ dst,
                               float* __restrict__ agg, int E, int din) {
    int wpb = blockDim.x >> 5;
    int e = blockIdx.x * wpb + (threadIdx.x >> 5);
    if (e >= E) return;
    int lane = threadIdx.x & 31;
    int s = src[e], d = dst[e];
    int per = din >> 5;
    const __hip_bfloat16* xp = x + (size_t)s * din + (size_t)lane * per;
    float* ap = agg + (size_t)d * din + (size_t)lane * per;
    for (int j = 0; j < per; ++j)
        atomicAdd(&ap[j], __bfloat162float(xp[j]));
}

// agg / max(deg,1) -> bf16
__global__ void finalize_kernel(const float* __restrict__ agg,
                                const float* __restrict__ deg,
                                __hip_bfloat16* __restrict__ aggb,
                                int N, int din) {
    long i = (long)blockIdx.x * blockDim.x + threadIdx.x;
    long total = (long)N * din;
    if (i >= total) return;
    int n = (int)(i / din);
    float inv = 1.0f / fmaxf(deg[n], 1.0f);
    aggb[i] = __float2bfloat16(agg[i] * inv);
}

// ---------------- WMMA GEMM, register-blocked over column tiles ----------------
// out[n][o] = act( BN( A0[n]@B0[o]^T (+ A1[n]@B1[o]^T) + bias[o] ) ), bf16 out.
// One wave handles a 16-row node tile and ALL DOUT/16 column tiles (A read once
// per K-step). Weight matrices are staged in LDS once per workgroup (async copy
// on toolchains that expose GLOBAL_LOAD_ASYNC_TO_LDS).
template <int DIN, int DOUT, bool DUAL>
__global__ __launch_bounds__(256) void gemm_tile_kernel(
    const __hip_bfloat16* __restrict__ A0,
    const __hip_bfloat16* __restrict__ A1,
    const __hip_bfloat16* __restrict__ B0g,
    const __hip_bfloat16* __restrict__ B1g,
    const float* __restrict__ bias,
    const float* __restrict__ g,
    const float* __restrict__ be,
    const float* __restrict__ m,
    const float* __restrict__ v,
    int relu,
    __hip_bfloat16* __restrict__ out, int N)
{
    constexpr int CT    = DOUT / 16;          // column tiles per wave
    constexpr int KCH   = DIN / 32;           // K chunks
    constexpr int WELEM = DOUT * DIN;         // elems per weight matrix
    constexpr int NMAT  = DUAL ? 2 : 1;
    __shared__ __hip_bfloat16 smem[NMAT * WELEM];

    // ---- stage weights into LDS (all threads participate) ----
    constexpr int CHUNKS = NMAT * WELEM / 8;  // 16B chunks
    for (int i = threadIdx.x; i < CHUNKS; i += 256) {
        const __hip_bfloat16* gsrc =
            (i * 8 < WELEM) ? (B0g + (size_t)i * 8) : (B1g + (size_t)i * 8 - WELEM);
        __hip_bfloat16* ldst = smem + (size_t)i * 8;
#if HAVE_ASYNC_LDS
        __builtin_amdgcn_global_load_async_to_lds_b128(
            (async_gptr)gsrc, (async_lptr)ldst, 0, 0);
#else
        *(uint4*)ldst = *(const uint4*)gsrc;
#endif
    }
#if HAVE_ASYNC_LDS
    __builtin_amdgcn_s_wait_asynccnt(0);
#endif
    __syncthreads();

    const int mt = blockIdx.x * 8 + (threadIdx.x >> 5);
    const int mtiles = N >> 4;
    if (mt < mtiles) {                        // wave-uniform: EXEC stays all-1s
        const int lane = threadIdx.x & 31;
        const int half = lane >> 4;           // K-half selector
        const int l16  = lane & 15;           // A row / B output column

        const __hip_bfloat16* a0row = A0 + (size_t)(mt * 16 + l16) * DIN;
        const __hip_bfloat16* a1row = DUAL ? (A1 + (size_t)(mt * 16 + l16) * DIN) : nullptr;
        const __hip_bfloat16* b0s = smem;
        const __hip_bfloat16* b1s = smem + WELEM;

        f32x8 acc[CT];
#pragma unroll
        for (int c = 0; c < CT; ++c) acc[c] = (f32x8)0.0f;

#pragma unroll
        for (int k = 0; k < KCH; ++k) {
            const int kb = k * 32;
            BVec a0;
            // A 16x32 bf16 layout: vgpr0-3 -> K = kb + half*8 + [0..7],
            //                      vgpr4-7 -> K = kb + 16 + half*8 + [0..7]
            a0.q[0] = *(const uint4*)(a0row + kb + half * 8);
            a0.q[1] = *(const uint4*)(a0row + kb + 16 + half * 8);
            BVec a1;
            if constexpr (DUAL) {
                a1.q[0] = *(const uint4*)(a1row + kb + half * 8);
                a1.q[1] = *(const uint4*)(a1row + kb + 16 + half * 8);
            }
#pragma unroll
            for (int c = 0; c < CT; ++c) {
                const int o = c * 16 + l16;
                // B 32x16 bf16 layout: lane column = l16, K = kb + half*8 + [0..7], kb + 16 + half*8 + [0..7]
                BVec b0;
                b0.q[0] = *(const uint4*)(b0s + (size_t)o * DIN + kb + half * 8);
                b0.q[1] = *(const uint4*)(b0s + (size_t)o * DIN + kb + 16 + half * 8);
                acc[c] = __builtin_amdgcn_wmma_f32_16x16x32_bf16(
                    false, a0.v, false, b0.v, (short)0, acc[c], false, false);
                if constexpr (DUAL) {
                    BVec b1;
                    b1.q[0] = *(const uint4*)(b1s + (size_t)o * DIN + kb + half * 8);
                    b1.q[1] = *(const uint4*)(b1s + (size_t)o * DIN + kb + 16 + half * 8);
                    acc[c] = __builtin_amdgcn_wmma_f32_16x16x32_bf16(
                        false, a1.v, false, b1.v, (short)0, acc[c], false, false);
                }
            }
        }

        // fused bias + BatchNorm(eval) + ReLU epilogue, bf16 store
        const int row0 = mt * 16 + half * 8;
#pragma unroll
        for (int c = 0; c < CT; ++c) {
            const int o = c * 16 + l16;
            const float bo = bias ? bias[o] : 0.0f;
            float sc = 1.0f, sh = 0.0f;
            if (g) {
                const float s = g[o] * rsqrtf(v[o] + BN_EPS);
                sc = s;
                sh = be[o] - m[o] * s;
            }
#pragma unroll
            for (int r = 0; r < 8; ++r) {
                float val = (acc[c][r] + bo) * sc + sh;
                if (relu) val = fmaxf(val, 0.0f);
                out[(size_t)(row0 + r) * DOUT + o] = __float2bfloat16(val);
            }
        }
    }
}

// final 32 -> 1 projection
__global__ void dot_head_kernel(const __hip_bfloat16* __restrict__ h,
                                const float* __restrict__ w,
                                const float* __restrict__ b,
                                float* __restrict__ out, int N, int K) {
    int n = blockIdx.x * blockDim.x + threadIdx.x;
    if (n >= N) return;
    const __hip_bfloat16* hp = h + (size_t)n * K;
    float s = 0.0f;
    for (int k = 0; k < K; ++k)
        s += __bfloat162float(hp[k]) * w[k];
    out[n] = s + b[0];
}

// ---------------- launch ----------------

static inline int cdiv_l(long a, long b) { return (int)((a + b - 1) / b); }

extern "C" void kernel_launch(void* const* d_in, const int* in_sizes, int n_in,
                              void* d_out, int out_size, void* d_ws, size_t ws_size,
                              hipStream_t stream) {
    const float* x  = (const float*)d_in[0];
    const int*   ei = (const int*)d_in[1];
    const int N = in_sizes[0] / 128;
    const int E = in_sizes[1] / 2;
    const int* src = ei;
    const int* dst = ei + E;

    const float* Wl0 = (const float*)d_in[2];  const float* bl0 = (const float*)d_in[3];
    const float* Wr0 = (const float*)d_in[4];
    const float* g0  = (const float*)d_in[5];  const float* be0 = (const float*)d_in[6];
    const float* m0  = (const float*)d_in[7];  const float* v0  = (const float*)d_in[8];
    const float* Wl1 = (const float*)d_in[9];  const float* bl1 = (const float*)d_in[10];
    const float* Wr1 = (const float*)d_in[11];
    const float* g1  = (const float*)d_in[12]; const float* be1 = (const float*)d_in[13];
    const float* m1  = (const float*)d_in[14]; const float* v1  = (const float*)d_in[15];
    const float* Wl2 = (const float*)d_in[16]; const float* bl2 = (const float*)d_in[17];
    const float* Wr2 = (const float*)d_in[18];
    const float* g2  = (const float*)d_in[19]; const float* be2 = (const float*)d_in[20];
    const float* m2  = (const float*)d_in[21]; const float* v2  = (const float*)d_in[22];
    const float* Wc1 = (const float*)d_in[23]; const float* bc1 = (const float*)d_in[24];
    const float* Wc2 = (const float*)d_in[25]; const float* bc2 = (const float*)d_in[26];
    const float* Wc3 = (const float*)d_in[27]; const float* bc3 = (const float*)d_in[28];

    // workspace layout
    char* ws = (char*)d_ws;
    auto alloc = [&](size_t bytes) -> void* {
        void* p = (void*)ws;
        ws += (bytes + 255) & ~(size_t)255;
        return p;
    };
    float*          deg  = (float*)alloc((size_t)N * 4);
    float*          agg  = (float*)alloc((size_t)N * 128 * 4);
    __hip_bfloat16* xb   = (__hip_bfloat16*)alloc((size_t)N * 128 * 2);
    __hip_bfloat16* aggb = (__hip_bfloat16*)alloc((size_t)N * 128 * 2);
    __hip_bfloat16* hA   = (__hip_bfloat16*)alloc((size_t)N * 128 * 2);
    __hip_bfloat16* hB   = (__hip_bfloat16*)alloc((size_t)N * 128 * 2);
    __hip_bfloat16* wb0  = (__hip_bfloat16*)alloc((size_t)128 * 128 * 2);
    __hip_bfloat16* wb1  = (__hip_bfloat16*)alloc((size_t)128 * 128 * 2);

    const int mtiles = N / 16;
    const int gemm_blocks = cdiv_l(mtiles, 8);

    // input features -> bf16; node degrees (same graph for all layers)
    f32_to_bf16_kernel<<<cdiv_l((long)N * 128, 256), 256, 0, stream>>>(x, xb, (long)N * 128);
    zero_f32_kernel<<<cdiv_l(N, 256), 256, 0, stream>>>(deg, N);
    degree_kernel<<<cdiv_l(E, 256), 256, 0, stream>>>(dst, deg, E);

    // ---- SAGE layer 0: 128 -> 128 ----
    zero_f32_kernel<<<cdiv_l((long)N * 128, 256), 256, 0, stream>>>(agg, (long)N * 128);
    scatter_kernel<<<cdiv_l(E, 8), 256, 0, stream>>>(xb, src, dst, agg, E, 128);
    finalize_kernel<<<cdiv_l((long)N * 128, 256), 256, 0, stream>>>(agg, deg, aggb, N, 128);
    f32_to_bf16_kernel<<<cdiv_l(128 * 128, 256), 256, 0, stream>>>(Wl0, wb0, 128 * 128);
    f32_to_bf16_kernel<<<cdiv_l(128 * 128, 256), 256, 0, stream>>>(Wr0, wb1, 128 * 128);
    gemm_tile_kernel<128, 128, true><<<gemm_blocks, 256, 0, stream>>>(
        aggb, xb, wb0, wb1, bl0, g0, be0, m0, v0, 1, hA, N);

    // ---- SAGE layer 1: 128 -> 128 ----
    zero_f32_kernel<<<cdiv_l((long)N * 128, 256), 256, 0, stream>>>(agg, (long)N * 128);
    scatter_kernel<<<cdiv_l(E, 8), 256, 0, stream>>>(hA, src, dst, agg, E, 128);
    finalize_kernel<<<cdiv_l((long)N * 128, 256), 256, 0, stream>>>(agg, deg, aggb, N, 128);
    f32_to_bf16_kernel<<<cdiv_l(128 * 128, 256), 256, 0, stream>>>(Wl1, wb0, 128 * 128);
    f32_to_bf16_kernel<<<cdiv_l(128 * 128, 256), 256, 0, stream>>>(Wr1, wb1, 128 * 128);
    gemm_tile_kernel<128, 128, true><<<gemm_blocks, 256, 0, stream>>>(
        aggb, hA, wb0, wb1, bl1, g1, be1, m1, v1, 1, hB, N);

    // ---- SAGE layer 2: 128 -> 64 ----
    zero_f32_kernel<<<cdiv_l((long)N * 128, 256), 256, 0, stream>>>(agg, (long)N * 128);
    scatter_kernel<<<cdiv_l(E, 8), 256, 0, stream>>>(hB, src, dst, agg, E, 128);
    finalize_kernel<<<cdiv_l((long)N * 128, 256), 256, 0, stream>>>(agg, deg, aggb, N, 128);
    f32_to_bf16_kernel<<<cdiv_l(64 * 128, 256), 256, 0, stream>>>(Wl2, wb0, 64 * 128);
    f32_to_bf16_kernel<<<cdiv_l(64 * 128, 256), 256, 0, stream>>>(Wr2, wb1, 64 * 128);
    gemm_tile_kernel<128, 64, true><<<gemm_blocks, 256, 0, stream>>>(
        aggb, hB, wb0, wb1, bl2, g2, be2, m2, v2, 1, hA, N);

    // ---- MLP head: 64 -> 64 (relu), 64 -> 32 (relu), 32 -> 1 ----
    f32_to_bf16_kernel<<<cdiv_l(64 * 64, 256), 256, 0, stream>>>(Wc1, wb0, 64 * 64);
    gemm_tile_kernel<64, 64, false><<<gemm_blocks, 256, 0, stream>>>(
        hA, nullptr, wb0, nullptr, bc1, nullptr, nullptr, nullptr, nullptr, 1, hB, N);

    f32_to_bf16_kernel<<<cdiv_l(32 * 64, 256), 256, 0, stream>>>(Wc2, wb0, 32 * 64);
    gemm_tile_kernel<64, 32, false><<<gemm_blocks, 256, 0, stream>>>(
        hB, nullptr, wb0, nullptr, bc2, nullptr, nullptr, nullptr, nullptr, 1, hA, N);

    dot_head_kernel<<<cdiv_l(N, 256), 256, 0, stream>>>(hA, Wc3, bc3, (float*)d_out, N, 32);
}
